// Net_55405078118821
// MI455X (gfx1250) — matrix operand order repaired and checked by reading, compiled, not run on hardware
//
#include <hip/hip_runtime.h>

typedef __attribute__((ext_vector_type(16))) _Float16 v16h;
typedef __attribute__((ext_vector_type(8)))  _Float16 v8h;
typedef __attribute__((ext_vector_type(8)))  float    v8f;
typedef __attribute__((ext_vector_type(4)))  int      v4i;
typedef __attribute__((address_space(1))) v4i as1_v4i;  // global
typedef __attribute__((address_space(3))) v4i as3_v4i;  // LDS

#define TSTEPS 16

// ---- gfx1250 async global->LDS copy (ASYNCcnt) with sync fallback ----
#if __has_builtin(__builtin_amdgcn_global_load_async_to_lds_b128) && \
    __has_builtin(__builtin_amdgcn_s_wait_asynccnt)
#define USE_ASYNC_COPY 1
#define WAIT_ASYNC() __builtin_amdgcn_s_wait_asynccnt(0)
#else
#define USE_ASYNC_COPY 0
#define WAIT_ASYNC() do {} while (0)
#endif

// LDS input tile: 7 rows x 20 cols (19 used) x 16 ch, f16, double buffered.
#define TROWS 7
#define TCOLS 20
#define TUSED 19
#define TILE_HALFS (TROWS * TCOLS * 16)

// ---------------- utility kernels ----------------

__global__ void zero_i32_kernel(int* p, int n) {
  int i = blockIdx.x * blockDim.x + threadIdx.x;
  if (i < n) p[i] = 0;
}

// Pre-swizzle conv weights [Co,16,4,4] (f32) into WMMA B fragments (f16).
// frag[j][lane][i] = B[k][n],  n = lane&15,  k = j*32 + (i>>3)*16 + (lane>>4)*8 + (i&7)
// B[k][n] = w[n][ci][kh][kw],  ci = k&15, tap = k>>4, kh = tap>>2, kw = tap&3
__global__ void prep_wfrag_kernel(const float* __restrict__ w2,
                                  const float* __restrict__ w3,
                                  const float* __restrict__ w4,
                                  const float* __restrict__ w5,
                                  _Float16* __restrict__ frag /* 4*4096 */) {
  int e = blockIdx.x * blockDim.x + threadIdx.x;
  if (e >= 4 * 4096) return;
  int L = e >> 12;
  int r = e & 4095;
  int j = r >> 9;
  int lane = (r >> 4) & 31;
  int i = r & 15;
  int n = lane & 15;
  int k = j * 32 + ((i >> 3) << 4) + ((lane >> 4) << 3) + (i & 7);
  int ci = k & 15, tap = k >> 4, kh = tap >> 2, kw = tap & 3;
  const float* w = (L == 0) ? w2 : (L == 1) ? w3 : (L == 2) ? w4 : w5;
  int Co = (L == 3) ? 6 : 16;
  float v = (n < Co) ? w[((n * 16 + ci) * 4 + kh) * 4 + kw] : 0.0f;
  frag[e] = (_Float16)v;
}

__global__ void finalize_out_kernel(const int* __restrict__ cnt,
                                    float* __restrict__ out, int n) {
  int i = blockIdx.x * blockDim.x + threadIdx.x;
  if (i < n) out[i] = (float)cnt[i] * (1.0f / (49.0f * 49.0f));
}

// ---------------- layer 1: direct conv (Cin=3) + LIF ----------------
// x: [32,16,3,64,64] f32 -> spikes [512][61][61][16] f16 (channels-last)
__global__ void snn_layer1_kernel(const float* __restrict__ x,
                                  const float* __restrict__ w1,
                                  const float* __restrict__ vth,
                                  _Float16* __restrict__ out) {
  __shared__ float ws[16][3][4][4];
  int tx = threadIdx.x;  // w within 16-wide tile
  int ty = threadIdx.y;  // co
  int tid = ty * 16 + tx;
  for (int i = tid; i < 16 * 3 * 16; i += 256) ((float*)ws)[i] = w1[i];
  __syncthreads();
  int w = blockIdx.x * 16 + tx;
  int h = blockIdx.y;
  int b = blockIdx.z;
  if (w >= 61) return;
  float vthv = vth[ty];
  float v = 0.0f;
  for (int t = 0; t < TSTEPS; ++t) {
    const float* xf = x + ((size_t)(b * TSTEPS + t)) * 3 * 64 * 64;
    float dv = 0.0f;
#pragma unroll
    for (int ci = 0; ci < 3; ++ci)
#pragma unroll
      for (int kh = 0; kh < 4; ++kh)
#pragma unroll
        for (int kw = 0; kw < 4; ++kw)
          dv += xf[(ci * 64 + (h + kh)) * 64 + (w + kw)] * ws[ty][ci][kh][kw];
    v += dv;
    float s = (v >= vthv) ? 1.0f : 0.0f;
    if (s != 0.0f) v = 0.0f;
    out[((((size_t)(b * TSTEPS + t)) * 61 + h) * 61 + w) * 16 + ty] = (_Float16)s;
  }
}

// ---------------- shared-tile staging ----------------
// Copy one timestep's 7x19 pixel tile (channels-last f16) into LDS.
// 266 16-byte chunks over 128 threads. Edge coords are clamped (duplicated
// data is only ever read by lanes/waves whose results are discarded).
__device__ __forceinline__ void copy_tile(const _Float16* __restrict__ inf,
                                          _Float16* __restrict__ lds,
                                          int Hi, int Wi, int h0, int w0,
                                          int tid) {
  for (int c = tid; c < TROWS * TUSED * 2; c += 128) {
    int half = c & 1;
    int p = c >> 1;
    int row = p / TUSED;
    int col = p - row * TUSED;
    int gr = h0 + row; if (gr > Hi - 1) gr = Hi - 1;
    int gc = w0 + col; if (gc > Wi - 1) gc = Wi - 1;
    const _Float16* g = inf + (((size_t)gr) * Wi + gc) * 16 + half * 8;
    _Float16* l = lds + (row * TCOLS + col) * 16 + half * 8;
#if USE_ASYNC_COPY
    __builtin_amdgcn_global_load_async_to_lds_b128(
        (as1_v4i*)g, (as3_v4i*)l, 0, 0);
#else
    *(v8h*)l = *(const v8h*)g;
#endif
  }
}

// A fragment (16x32 f16) for K-chunk j from the LDS tile.
// Lane covers pixel m=lane&15 at tile column mcol; lane half selects ci 0-7/8-15.
__device__ __forceinline__ v16h load_a_frag_lds(const _Float16* __restrict__ tile,
                                                int dh, int mcol, int cib, int j) {
  int tap0 = 2 * j, tap1 = 2 * j + 1;
  int kh0 = tap0 >> 2, kw0 = tap0 & 3;
  int kh1 = tap1 >> 2, kw1 = tap1 & 3;
  v8h a0 = *(const v8h*)(tile + ((dh + kh0) * TCOLS + (mcol + kw0)) * 16 + cib);
  v8h a1 = *(const v8h*)(tile + ((dh + kh1) * TCOLS + (mcol + kw1)) * 16 + cib);
  v16h a;
#pragma unroll
  for (int i = 0; i < 8; ++i) { a[i] = a0[i]; a[i + 8] = a1[i]; }
  return a;
}

// ---------------- layers 2-4: LDS-staged WMMA conv + LIF ----------------
// 128 threads = 4 waves = 4 consecutive output rows sharing one input tile.
__global__ void __launch_bounds__(128)
snn_wmma_layer_kernel(const _Float16* __restrict__ in,   // [512][Hi][Wi][16]
                      const _Float16* __restrict__ wfrag, // [8][32][16] f16
                      const float* __restrict__ vth,      // [16]
                      _Float16* __restrict__ out,         // [512][Ho][Wo][16]
                      int Hi, int Wi, int Ho, int Wo) {
  __shared__ _Float16 tile[2][TILE_HALFS];
  int tid = threadIdx.x;
  int lane = tid & 31;
  int dh = tid >> 5;      // wave index = output row offset in tile
  int m = lane & 15;
  int hilo = lane >> 4;
  int w0 = blockIdx.x * 16;
  int h0 = blockIdx.y * 4;
  int h = h0 + dh;
  int b = blockIdx.z;
  bool hvalid = (h < Ho);

  v16h bf[8];
#pragma unroll
  for (int j = 0; j < 8; ++j)
    bf[j] = *(const v16h*)(wfrag + (j * 32 + lane) * 16);

  int mcol = m; if (w0 + mcol > Wo - 1) mcol = Wo - 1 - w0;  // EXEC stays full
  int cib = hilo * 8;
  float vthv = vth[m];
  v8f v = {};  // membrane: reg r <-> pixel w0+r+8*hilo, channel m

  const size_t frame = (size_t)Hi * Wi * 16;
  const _Float16* base = in + (size_t)b * TSTEPS * frame;

  copy_tile(base, tile[0], Hi, Wi, h0, w0, tid);
  for (int t = 0; t < TSTEPS; ++t) {
    WAIT_ASYNC();
    __syncthreads();
    if (t + 1 < TSTEPS)
      copy_tile(base + (size_t)(t + 1) * frame, tile[(t + 1) & 1],
                Hi, Wi, h0, w0, tid);
    const _Float16* cur = tile[t & 1];
#pragma unroll
    for (int j = 0; j < 8; ++j) {
      v16h a = load_a_frag_lds(cur, dh, mcol, cib, j);
      v = __builtin_amdgcn_wmma_f32_16x16x32_f16(false, a, false, bf[j],
                                                 (short)0, v, false, false);
    }
    _Float16* of = out + ((((size_t)(b * TSTEPS + t)) * Ho + h) * Wo) * 16;
#pragma unroll
    for (int r = 0; r < 8; ++r) {
      int wp = w0 + r + 8 * hilo;
      float s = (v[r] >= vthv) ? 1.0f : 0.0f;
      if (s != 0.0f) v[r] = 0.0f;
      if (hvalid && wp < Wo) of[(size_t)wp * 16 + m] = (_Float16)s;
    }
  }
}

// ---------------- layer 5: LDS-staged WMMA + LIF + spatial-mean ----------------
// Cout=6 (B padded with zeros). Integer spike counts accumulated atomically
// (exact & deterministic), scaled afterwards.
__global__ void __launch_bounds__(128)
snn_wmma_final_kernel(const _Float16* __restrict__ in,
                      const _Float16* __restrict__ wfrag,
                      const float* __restrict__ vth,  // [6]
                      int* __restrict__ cnt,          // [32*16*6]
                      int Hi, int Wi, int Ho, int Wo) {
  __shared__ _Float16 tile[2][TILE_HALFS];
  int tid = threadIdx.x;
  int lane = tid & 31;
  int dh = tid >> 5;
  int m = lane & 15;
  int hilo = lane >> 4;
  int w0 = blockIdx.x * 16;
  int h0 = blockIdx.y * 4;
  int h = h0 + dh;
  int b = blockIdx.z;
  bool hvalid = (h < Ho);

  v16h bf[8];
#pragma unroll
  for (int j = 0; j < 8; ++j)
    bf[j] = *(const v16h*)(wfrag + (j * 32 + lane) * 16);

  int mcol = m; if (w0 + mcol > Wo - 1) mcol = Wo - 1 - w0;
  int cib = hilo * 8;
  float vthv = (m < 6) ? vth[m] : 3.0e38f;  // padded channels never fire
  v8f v = {};

  const size_t frame = (size_t)Hi * Wi * 16;
  const _Float16* base = in + (size_t)b * TSTEPS * frame;

  copy_tile(base, tile[0], Hi, Wi, h0, w0, tid);
  for (int t = 0; t < TSTEPS; ++t) {
    WAIT_ASYNC();
    __syncthreads();
    if (t + 1 < TSTEPS)
      copy_tile(base + (size_t)(t + 1) * frame, tile[(t + 1) & 1],
                Hi, Wi, h0, w0, tid);
    const _Float16* cur = tile[t & 1];
#pragma unroll
    for (int j = 0; j < 8; ++j) {
      v16h a = load_a_frag_lds(cur, dh, mcol, cib, j);
      v = __builtin_amdgcn_wmma_f32_16x16x32_f16(false, a, false, bf[j],
                                                 (short)0, v, false, false);
    }
    int sum = 0;
#pragma unroll
    for (int r = 0; r < 8; ++r) {
      int wp = w0 + r + 8 * hilo;
      float s = (v[r] >= vthv) ? 1.0f : 0.0f;
      if (s != 0.0f) v[r] = 0.0f;
      if (wp < Wo) sum += (int)s;
    }
    sum += __shfl_xor(sum, 16, 32);  // combine pixel halves of same channel
    if (hvalid && lane < 6)
      atomicAdd(&cnt[(b * TSTEPS + t) * 6 + lane], sum);
  }
}

// ---------------- launcher ----------------
extern "C" void kernel_launch(void* const* d_in, const int* in_sizes, int n_in,
                              void* d_out, int out_size, void* d_ws, size_t ws_size,
                              hipStream_t stream) {
  (void)in_sizes; (void)n_in; (void)ws_size;
  const float* x    = (const float*)d_in[0];
  const float* w1   = (const float*)d_in[1];
  const float* vth1 = (const float*)d_in[2];
  const float* w2   = (const float*)d_in[3];
  const float* vth2 = (const float*)d_in[4];
  const float* w3   = (const float*)d_in[5];
  const float* vth3 = (const float*)d_in[6];
  const float* w4   = (const float*)d_in[7];
  const float* vth4 = (const float*)d_in[8];
  const float* w5   = (const float*)d_in[9];
  const float* vth5 = (const float*)d_in[10];
  float* out = (float*)d_out;

  char* ws = (char*)d_ws;
  _Float16* wfrag = (_Float16*)ws;                         // 4*4096 f16 = 32 KB
  int* cnt = (int*)(ws + 32768);                           // 3072 ints
  size_t offA = 32768 + 3072 * sizeof(int);                // 45056 (32B aligned)
  size_t sizeA = (size_t)512 * 61 * 61 * 16 * 2;           // ~58.1 MB (max of L1/L3 outs)
  _Float16* bufA = (_Float16*)(ws + offA);
  _Float16* bufB = (_Float16*)(ws + offA + sizeA);         // ~52.6 MB (max of L2/L4 outs)

  zero_i32_kernel<<<(out_size + 255) / 256, 256, 0, stream>>>(cnt, out_size);
  prep_wfrag_kernel<<<(4 * 4096 + 255) / 256, 256, 0, stream>>>(w2, w3, w4, w5, wfrag);

  // L1: x -> bufA [512][61][61][16]
  snn_layer1_kernel<<<dim3(4, 61, 32), dim3(16, 16), 0, stream>>>(x, w1, vth1, bufA);
  // L2: bufA(61x61) -> bufB(58x58)
  snn_wmma_layer_kernel<<<dim3(4, (58 + 3) / 4, 32), 128, 0, stream>>>(
      bufA, wfrag + 0 * 4096, vth2, bufB, 61, 61, 58, 58);
  // L3: bufB(58x58) -> bufA(55x55)
  snn_wmma_layer_kernel<<<dim3(4, (55 + 3) / 4, 32), 128, 0, stream>>>(
      bufB, wfrag + 1 * 4096, vth3, bufA, 58, 58, 55, 55);
  // L4: bufA(55x55) -> bufB(52x52)
  snn_wmma_layer_kernel<<<dim3(4, (52 + 3) / 4, 32), 128, 0, stream>>>(
      bufA, wfrag + 2 * 4096, vth4, bufB, 55, 55, 52, 52);
  // L5: bufB(52x52) -> spike counts (49x49 outputs)
  snn_wmma_final_kernel<<<dim3(4, (49 + 3) / 4, 32), 128, 0, stream>>>(
      bufB, wfrag + 3 * 4096, vth5, cnt, 52, 52, 49, 49);
  // mean = count / 49^2
  finalize_out_kernel<<<(out_size + 255) / 256, 256, 0, stream>>>(cnt, out, out_size);
}
